// LSTM_AE_18863496364499
// MI455X (gfx1250) — matrix operand order repaired and checked by reading, not compiled, not run on hardware
//
#include <hip/hip_runtime.h>
#include <math.h>

// ---------------------------------------------------------------------------
// LSTM autoencoder, MI455X (gfx1250), wave32 + v_wmma_f32_16x16x32_bf16.
// B=256, T=512, F=76 (K padded to 96), H=128 (4H=512 gates).
// ---------------------------------------------------------------------------

#define BB   256
#define TT   512
#define FF   76
#define HH   128
#define NG   512          // 4*H gate width
#define KCX  3            // k-chunks for x/din side (96 = 3*32)
#define KCH  4            // k-chunks for h side (128 = 4*32)
#define KCFC 8            // k-chunks for FC (256 = 8*32)
#define NTG  32           // n-tiles for gates (512/16)
#define NTFC 5            // n-tiles for FC (80/16, cols >=76 masked)

typedef __attribute__((ext_vector_type(16))) __bf16 v16bf;
typedef __attribute__((ext_vector_type(8)))  float  v8f;

// Defeat LICM of loop-invariant weight-fragment loads WITHOUT destroying the
// pointer's global-address-space provenance: launder a zero SGPR *offset* and
// add it to the (still provably global) base pointer. Loads stay per-iteration
// global_load_b128 (saddr + imm), hitting L2; no hoist -> no scratch spills,
// and no flat_load fallback.
#define LAUNDER_OFF(o) asm volatile("" : "+s"(o))

__device__ __forceinline__ v16bf ld16(const __bf16* p) {
  return *reinterpret_cast<const v16bf*>(p);
}
__device__ __forceinline__ v8f wmma_bf16(v16bf a, v16bf b, v8f c) {
  return __builtin_amdgcn_wmma_f32_16x16x32_bf16(false, a, false, b, (short)0, c,
                                                 false, false);
}
__device__ __forceinline__ float sigm(float x) { return 1.0f / (1.0f + __expf(-x)); }

// A-fragment slot -> kk  (16-bit A 16x32 layout, ISA 7.12.2)
__device__ __forceinline__ int a_kk(int slot, int laneHi) {
  return (slot & 7) + (laneHi << 3) + ((slot >> 3) << 4);
}

// ---------------------------------------------------------------------------
// Pack x[B,T,F] f32 -> bf16 A-fragments: [b_tile(16)][t(512)][kc(3)][512]
// ---------------------------------------------------------------------------
__global__ __launch_bounds__(256) void pack_x_kernel(const float* __restrict__ x,
                                                     __bf16* __restrict__ xp) {
  size_t idx = (size_t)blockIdx.x * 256 + threadIdx.x;
  const size_t total = (size_t)16 * TT * KCX * 512;
  if (idx >= total) return;
  int ls = idx & 511;
  size_t rest = idx >> 9;
  int kc = rest % KCX; rest /= KCX;
  int t  = rest % TT;
  int bt = rest / TT;
  int ln = ls >> 4, slot = ls & 15;
  int r = ln & 15, lhi = ln >> 4;
  int k = kc * 32 + a_kk(slot, lhi);
  int b = bt * 16 + r;
  float v = (k < FF) ? x[((size_t)b * TT + t) * FF + k] : 0.0f;
  xp[idx] = (__bf16)v;
}

// ---------------------------------------------------------------------------
// Pack W[N,K] f32 (row-major) -> bf16 B-fragments: [nt][kc][512]
// B 32x16 layout: kk = (slot&7) + 8*(lane>=16) + 16*(slot>>3) (matches A).
// ---------------------------------------------------------------------------
__global__ __launch_bounds__(256) void pack_w_kernel(const float* __restrict__ W,
                                                     __bf16* __restrict__ dst,
                                                     int N, int K, int NT, int KC) {
  int idx = blockIdx.x * 256 + threadIdx.x;
  int total = NT * KC * 512;
  if (idx >= total) return;
  int ls = idx & 511;
  int rest = idx >> 9;
  int kc = rest % KC;
  int nt = rest / KC;
  int ln = ls >> 4, slot = ls & 15;
  int n = nt * 16 + (ln & 15);
  int k = kc * 32 + (slot & 7) + ((ln >> 4) << 3) + ((slot >> 3) << 4);
  float v = (n < N && k < K) ? W[(size_t)n * K + k] : 0.0f;
  dst[idx] = (__bf16)v;
}

// ---------------------------------------------------------------------------
// Encoder: 16 blocks x 256 threads (8 waves). Block owns 16 batch rows and
// scans t=0..511. Per step per cell: 4 n-tiles/wave, 7 WMMA per tile.
// h,c accumulate in f32 registers; h re-quantized to bf16 A-fragment in LDS.
// ---------------------------------------------------------------------------
__global__ __launch_bounds__(256) void enc_kernel(
    const __bf16* __restrict__ xpack,   // [16][512][3][512]
    const __bf16* __restrict__ wW,      // [2][32][3][512]
    const __bf16* __restrict__ wU,      // [2][32][4][512]
    const float* __restrict__ bias0,    // ebf [512]
    const float* __restrict__ bias1,    // ebb [512]
    float* __restrict__ gates,          // [16][2][16][512]
    float* __restrict__ enc_h)          // [2][256][128]
{
  __shared__ __align__(32) __bf16 hfrag[2][KCH][512];   // 8 KB

  const int blk  = blockIdx.x;
  const int tid  = threadIdx.x;
  const int lane = tid & 31;
  const int wave = tid >> 5;
  const int hi8  = (lane >> 4) << 3;

  for (int i = tid; i < 2 * KCH * 512; i += 256)
    ((__bf16*)hfrag)[i] = (__bf16)0.0f;

  // update-phase ownership: 128 threads per cell, one hidden column each
  const int ucell = tid >> 7;
  const int uj    = tid & 127;
  float c_acc[16], h_acc[16];
#pragma unroll
  for (int r = 0; r < 16; ++r) { c_acc[r] = 0.0f; h_acc[r] = 0.0f; }

  float* gbase = gates + (size_t)blk * (2 * 16 * NG);
  __syncthreads();

  for (int t = 0; t < TT; ++t) {
    // opaque per-iteration zero offset: keeps weight loads inside the loop
    size_t zoff = 0;
    LAUNDER_OFF(zoff);
    const __bf16* wWt = wW + zoff;
    const __bf16* wUt = wU + zoff;

    // ---- gate GEMMs ----
    const __bf16* xb = xpack + (((size_t)blk * TT + t) * KCX) * 512 + lane * 16;
    v16bf xa0 = ld16(xb);
    v16bf xa1 = ld16(xb + 512);
    v16bf xa2 = ld16(xb + 1024);

#pragma unroll
    for (int cell = 0; cell < 2; ++cell) {
      v16bf ha0 = ld16(&hfrag[cell][0][lane * 16]);
      v16bf ha1 = ld16(&hfrag[cell][1][lane * 16]);
      v16bf ha2 = ld16(&hfrag[cell][2][lane * 16]);
      v16bf ha3 = ld16(&hfrag[cell][3][lane * 16]);
#pragma unroll
      for (int ntl = 0; ntl < 4; ++ntl) {
        int nt = wave * 4 + ntl;
        const __bf16* pw = wWt + ((size_t)(cell * NTG + nt) * KCX) * 512 + lane * 16;
        const __bf16* pu = wUt + ((size_t)(cell * NTG + nt) * KCH) * 512 + lane * 16;
        v8f acc = {};
        acc = wmma_bf16(xa0, ld16(pw),        acc);
        acc = wmma_bf16(xa1, ld16(pw + 512),  acc);
        acc = wmma_bf16(xa2, ld16(pw + 1024), acc);
        acc = wmma_bf16(ha0, ld16(pu),        acc);
        acc = wmma_bf16(ha1, ld16(pu + 512),  acc);
        acc = wmma_bf16(ha2, ld16(pu + 1024), acc);
        acc = wmma_bf16(ha3, ld16(pu + 1536), acc);
        float* gp = gbase + (size_t)cell * 16 * NG + nt * 16 + (lane & 15);
#pragma unroll
        for (int i = 0; i < 8; ++i) gp[(size_t)(i + hi8) * NG] = acc[i];
      }
    }
    __threadfence();
    __syncthreads();

    // ---- state update (accumulating states, as in reference) ----
    {
      const float* bb = ucell ? bias1 : bias0;
      const float* g0 = gbase + (size_t)ucell * 16 * NG;
      float bi = bb[uj], bf = bb[128 + uj], bg = bb[256 + uj], bo = bb[384 + uj];
#pragma unroll
      for (int r = 0; r < 16; ++r) {
        float iv = g0[(size_t)r * NG + uj] + bi;
        float fv = g0[(size_t)r * NG + 128 + uj] + bf;
        float gv = g0[(size_t)r * NG + 256 + uj] + bg;
        float ov = g0[(size_t)r * NG + 384 + uj] + bo;
        float c2 = sigm(fv) * c_acc[r] + sigm(iv) * tanhf(gv);
        float h2 = sigm(ov) * tanhf(c2);
        c_acc[r] += c2;
        h_acc[r] += h2;
        int kk = uj & 31, ch = uj >> 5;
        int slot = (kk & 7) + ((kk >> 4) << 3);
        int lhi  = (kk >> 3) & 1;
        hfrag[ucell][ch][(r + 16 * lhi) * 16 + slot] = (__bf16)h_acc[r];
      }
    }
    __threadfence();
    __syncthreads();
  }

#pragma unroll
  for (int r = 0; r < 16; ++r)
    enc_h[((size_t)ucell * BB + blk * 16 + r) * HH + uj] = h_acc[r];
}

// ---------------------------------------------------------------------------
// Decoder: 1 block x 512 threads (16 waves -> 2x the per-wave VGPR budget of a
// 1024-thread block; avoids WMMA-operand spills). Batch-coupled argmax-union
// feedback forces the per-step sync, done with workgroup barriers.
// Gate phase: each wave runs 2 (cell, m_tile) combos; 32 n-tiles; 7 WMMA/tile.
// ---------------------------------------------------------------------------
__global__ __launch_bounds__(512) void dec_kernel(
    const __bf16* __restrict__ wW,     // [2][32][3][512]
    const __bf16* __restrict__ wU,     // [2][32][4][512]
    const __bf16* __restrict__ fcp,    // [5][8][512]
    const float* __restrict__ dbf, const float* __restrict__ dbb,
    const float* __restrict__ fcb,
    const float* __restrict__ enc_h,   // [2][256][128]
    float* __restrict__ hstate,        // [2][256][128]
    float* __restrict__ cstate,        // [2][256][128]
    float* __restrict__ gates,         // [2][256][512]
    __bf16* __restrict__ hfragG,       // [2][16][4][512]
    __bf16* __restrict__ dinfrag,      // [16][3][512]
    float* __restrict__ out)           // [256][512][76]
{
  __shared__ float flags[6][16];
  const int tid  = threadIdx.x;
  const int lane = tid & 31;
  const int wave = tid >> 5;
  const int hi8  = (lane >> 4) << 3;
  const int BEG6[6] = {2, 10, 20, 30, 40, 50};
  const int END6[6] = {10, 20, 30, 40, 50, 59};

  // ---- init: h from encoder, c = 0, din = 0 ----
  for (int i = tid; i < 2 * BB * HH; i += 512) {
    int cl = i >> 15, row = (i >> 7) & 255, j = i & 127;
    float hv = enc_h[i];
    hstate[i] = hv;
    cstate[i] = 0.0f;
    int m = row >> 4, r = row & 15;
    int kk = j & 31, ch = j >> 5;
    int slot = (kk & 7) + ((kk >> 4) << 3);
    int lhi  = (kk >> 3) & 1;
    hfragG[((size_t)(cl * 16 + m) * KCH + ch) * 512 + (r + 16 * lhi) * 16 + slot] =
        (__bf16)hv;
  }
  for (int i = tid; i < 16 * KCX * 512; i += 512) dinfrag[i] = (__bf16)0.0f;
  __threadfence();
  __syncthreads();

  for (int t = 0; t < TT; ++t) {
    // ---- phase 1: gate GEMMs (2 combos per wave) ----
    {
      size_t zoff = 0;
      LAUNDER_OFF(zoff);
      const __bf16* wWt = wW + zoff;
      const __bf16* wUt = wU + zoff;
      for (int sub = 0; sub < 2; ++sub) {
        int combo = wave * 2 + sub;
        int cl = combo >> 4, m = combo & 15;
        const __bf16* pd = dinfrag + ((size_t)m * KCX) * 512 + lane * 16;
        v16bf ad0 = ld16(pd), ad1 = ld16(pd + 512), ad2 = ld16(pd + 1024);
        const __bf16* ph = hfragG + ((size_t)(cl * 16 + m) * KCH) * 512 + lane * 16;
        v16bf ah0 = ld16(ph), ah1 = ld16(ph + 512), ah2 = ld16(ph + 1024),
              ah3 = ld16(ph + 1536);
        for (int nt = 0; nt < NTG; ++nt) {
          const __bf16* pw = wWt + ((size_t)(cl * NTG + nt) * KCX) * 512 + lane * 16;
          const __bf16* pu = wUt + ((size_t)(cl * NTG + nt) * KCH) * 512 + lane * 16;
          v8f acc = {};
          acc = wmma_bf16(ad0, ld16(pw),        acc);
          acc = wmma_bf16(ad1, ld16(pw + 512),  acc);
          acc = wmma_bf16(ad2, ld16(pw + 1024), acc);
          acc = wmma_bf16(ah0, ld16(pu),        acc);
          acc = wmma_bf16(ah1, ld16(pu + 512),  acc);
          acc = wmma_bf16(ah2, ld16(pu + 1024), acc);
          acc = wmma_bf16(ah3, ld16(pu + 1536), acc);
          float* gp = gates + (size_t)(cl * BB + m * 16) * NG + nt * 16 + (lane & 15);
#pragma unroll
          for (int i = 0; i < 8; ++i) gp[(size_t)(i + hi8) * NG] = acc[i];
        }
      }
    }
    __threadfence();
    __syncthreads();

    // ---- phase 2: state update ----
    for (int i = tid; i < 2 * BB * HH; i += 512) {
      int cl = i >> 15, row = (i >> 7) & 255, j = i & 127;
      const float* bb = cl ? dbb : dbf;
      const float* g0 = gates + (size_t)(cl * BB + row) * NG;
      float iv = g0[j] + bb[j];
      float fv = g0[128 + j] + bb[128 + j];
      float gv = g0[256 + j] + bb[256 + j];
      float ov = g0[384 + j] + bb[384 + j];
      float cv = cstate[i];
      float c2 = sigm(fv) * cv + sigm(iv) * tanhf(gv);
      float h2 = sigm(ov) * tanhf(c2);
      cstate[i] = cv + c2;
      float hv2 = hstate[i] + h2;
      hstate[i] = hv2;
      int m = row >> 4, r = row & 15;
      int kk = j & 31, ch = j >> 5;
      int slot = (kk & 7) + ((kk >> 4) << 3);
      int lhi  = (kk >> 3) & 1;
      hfragG[((size_t)(cl * 16 + m) * KCH + ch) * 512 + (r + 16 * lhi) * 16 + slot] =
          (__bf16)hv2;
    }
    __threadfence();
    __syncthreads();

    // ---- phase 3: FC head (all 16 waves; K=256 = concat(hf,hb)) ----
    {
      size_t zoff = 0;
      LAUNDER_OFF(zoff);
      const __bf16* fct = fcp + zoff;
      int m = wave;
      v16bf a[KCFC];
#pragma unroll
      for (int kc = 0; kc < KCFC; ++kc)
        a[kc] = ld16(hfragG + ((size_t)((kc >> 2) * 16 + m) * KCH + (kc & 3)) * 512 +
                     lane * 16);
#pragma unroll
      for (int nt = 0; nt < NTFC; ++nt) {
        const __bf16* pf = fct + ((size_t)nt * KCFC) * 512 + lane * 16;
        v8f acc = {};
#pragma unroll
        for (int kc = 0; kc < KCFC; ++kc)
          acc = wmma_bf16(a[kc], ld16(pf + kc * 512), acc);
        int f = nt * 16 + (lane & 15);
        if (f < FF) {
          float bfc = fcb[f];
#pragma unroll
          for (int i = 0; i < 8; ++i) {
            int b = m * 16 + i + hi8;
            out[((size_t)b * TT + t) * FF + f] = acc[i] + bfc;
          }
        }
      }
    }
    __threadfence();
    __syncthreads();

    // ---- phase 4: argmax union -> one-hot columns -> next din fragment ----
    if (tid < 96) ((float*)flags)[tid] = 0.0f;
    __syncthreads();
    if (tid < BB) {
      const float* orow = out + ((size_t)tid * TT + t) * FF;
      for (int b = 0; b < 6; ++b) {
        int b0 = BEG6[b], e0 = END6[b];
        float mx = orow[b0];
        int am = 0;
        for (int k = b0 + 1; k < e0; ++k) {
          float v = orow[k];
          if (v > mx) { mx = v; am = k - b0; }
        }
        flags[b][am] = 1.0f;   // races write identical 1.0f -> benign
      }
    }
    __syncthreads();
    for (int e = tid; e < 16 * KCX * 512; e += 512) {
      int ls = e & 511;
      int rest = e >> 9;
      int kc = rest % KCX;
      int m  = rest / KCX;
      int ln = ls >> 4, slot = ls & 15;
      int r = ln & 15, lhi = ln >> 4;
      int k = kc * 32 + a_kk(slot, lhi);
      float val = 0.0f;
      if (k < FF) {
        if (k >= 2 && k < 59) {
          int blk = 0;
          for (int b = 0; b < 6; ++b)
            if (k >= BEG6[b] && k < END6[b]) blk = b;
          val = flags[blk][k - BEG6[blk]];
        } else {
          int b = m * 16 + r;
          val = out[((size_t)b * TT + t) * FF + k];
        }
      }
      dinfrag[e] = (__bf16)val;
    }
    __threadfence();
    __syncthreads();
  }
}

// ---------------------------------------------------------------------------
// Host side
// ---------------------------------------------------------------------------
extern "C" void kernel_launch(void* const* d_in, const int* in_sizes, int n_in,
                              void* d_out, int out_size, void* d_ws, size_t ws_size,
                              hipStream_t stream) {
  (void)in_sizes; (void)n_in; (void)out_size; (void)ws_size;
  const float* x   = (const float*)d_in[0];
  const float* eWf = (const float*)d_in[1];
  const float* eUf = (const float*)d_in[2];
  const float* ebf = (const float*)d_in[3];
  const float* eWb = (const float*)d_in[4];
  const float* eUb = (const float*)d_in[5];
  const float* ebb = (const float*)d_in[6];
  const float* dWf = (const float*)d_in[7];
  const float* dUf = (const float*)d_in[8];
  const float* dbf = (const float*)d_in[9];
  const float* dWb = (const float*)d_in[10];
  const float* dUb = (const float*)d_in[11];
  const float* dbb = (const float*)d_in[12];
  const float* fcW = (const float*)d_in[13];
  const float* fcb = (const float*)d_in[14];
  float* out = (float*)d_out;

  char* ws = (char*)d_ws;
  size_t off = 0;
  auto alloc = [&](size_t bytes) {
    char* p = ws + off;
    off += (bytes + 255) & ~(size_t)255;
    return p;
  };
  __bf16* xpack = (__bf16*)alloc((size_t)16 * TT * KCX * 512 * 2);   // ~24 MB
  __bf16* wWe   = (__bf16*)alloc((size_t)2 * NTG * KCX * 512 * 2);
  __bf16* wUe   = (__bf16*)alloc((size_t)2 * NTG * KCH * 512 * 2);
  __bf16* wWd   = (__bf16*)alloc((size_t)2 * NTG * KCX * 512 * 2);
  __bf16* wUd   = (__bf16*)alloc((size_t)2 * NTG * KCH * 512 * 2);
  __bf16* fcp   = (__bf16*)alloc((size_t)NTFC * KCFC * 512 * 2);
  float*  ench  = (float*)alloc((size_t)2 * BB * HH * 4);
  float*  gatE  = (float*)alloc((size_t)16 * 2 * 16 * NG * 4);
  float*  hst   = (float*)alloc((size_t)2 * BB * HH * 4);
  float*  cst   = (float*)alloc((size_t)2 * BB * HH * 4);
  float*  gatD  = (float*)alloc((size_t)2 * BB * NG * 4);
  __bf16* hfg   = (__bf16*)alloc((size_t)2 * 16 * KCH * 512 * 2);
  __bf16* dinf  = (__bf16*)alloc((size_t)16 * KCX * 512 * 2);

  // pack x
  {
    size_t total = (size_t)16 * TT * KCX * 512;
    pack_x_kernel<<<dim3((total + 255) / 256), dim3(256), 0, stream>>>(x, xpack);
  }
  // pack weights (B-fragments)
  auto packw = [&](const float* W, __bf16* dst, int N, int K, int NT, int KC) {
    int total = NT * KC * 512;
    pack_w_kernel<<<dim3((total + 255) / 256), dim3(256), 0, stream>>>(W, dst, N, K,
                                                                       NT, KC);
  };
  packw(eWf, wWe,                     NG, FF,  NTG, KCX);
  packw(eWb, wWe + NTG * KCX * 512,   NG, FF,  NTG, KCX);
  packw(eUf, wUe,                     NG, HH,  NTG, KCH);
  packw(eUb, wUe + NTG * KCH * 512,   NG, HH,  NTG, KCH);
  packw(dWf, wWd,                     NG, FF,  NTG, KCX);
  packw(dWb, wWd + NTG * KCX * 512,   NG, FF,  NTG, KCX);
  packw(dUf, wUd,                     NG, HH,  NTG, KCH);
  packw(dUb, wUd + NTG * KCH * 512,   NG, HH,  NTG, KCH);
  packw(fcW, fcp,                     FF, 2 * HH, NTFC, KCFC);

  // encoder scan: 16 blocks, each owns 16 batch rows
  enc_kernel<<<dim3(16), dim3(256), 0, stream>>>(xpack, wWe, wUe, ebf, ebb, gatE,
                                                 ench);
  // decoder scan: single persistent workgroup (16 waves)
  dec_kernel<<<dim3(1), dim3(512), 0, stream>>>(wWd, wUd, fcp, dbf, dbb, fcb, ench,
                                                hst, cst, gatD, hfg, dinf, out);
}
